// Lstm_model_88356067213889
// MI455X (gfx1250) — matrix operand order repaired and checked by reading, not compiled, not run on hardware
//
#include <hip/hip_runtime.h>
#include <hip/hip_bf16.h>

// ---------------------------------------------------------------------------
// CDNA5 (gfx1250) batch-parallel bidirectional LSTM.
//
// Transposed gate formulation:  gates^T[4H x 16b] = W[4H x K] * [x;h]^T[K x 16b]
//   - A-matrix (v16h)  = weight tile, resident in VGPRs for the whole T loop
//   - B-matrix (v16h)  = x^T / h^T tile, lane = batch column
//   - C/D      (v8f)   = gates^T tile, lane = batch column, VGPR = gate row
// D-layout -> next-step B-layout conversion is per-lane (+ one shfl_xor(16)).
// ---------------------------------------------------------------------------

typedef __attribute__((ext_vector_type(16))) _Float16 v16h;
typedef __attribute__((ext_vector_type(8)))  float    v8f;

__device__ __forceinline__ v8f wmma_f16f32(v16h a, v16h b, v8f c) {
  // (neg_a, A, neg_b, B, c_mod, C, reuse_a, reuse_b)
  return __builtin_amdgcn_wmma_f32_16x16x32_f16(false, a, false, b, (short)0, c,
                                                false, false);
}

__device__ __forceinline__ float sigm(float x) {
  x = fminf(fmaxf(x, -30.f), 30.f);
  return __builtin_amdgcn_rcpf(1.f + __expf(-x));
}
__device__ __forceinline__ float tanh_fast(float x) {
  x = fminf(fmaxf(x, -15.f), 15.f);
  float e = __expf(2.f * x);
  return (e - 1.f) * __builtin_amdgcn_rcpf(e + 1.f);
}

// I: input width (multiple of 32). H: true hidden. Hp: padded hidden (16|32).
template <int I, int H, int Hp>
__global__ __launch_bounds__(256)
void lstm_layer_kernel(const float* __restrict__ x,     // [B, T, I]
                       float* __restrict__ out,          // [B, T, outCols]
                       int outCols,
                       const float* __restrict__ WihF, const float* __restrict__ WhhF,
                       const float* __restrict__ bihF, const float* __restrict__ bhhF,
                       const float* __restrict__ WihB, const float* __restrict__ WhhB,
                       const float* __restrict__ bihB, const float* __restrict__ bhhB,
                       int B, int T) {
  constexpr int NT  = (4 * Hp) / 16;  // gate M-tiles
  constexpr int CT  = Hp / 16;        // hidden tiles (1 or 2)
  constexpr int KTX = I / 32;         // K-tiles for the input projection

  const int lane   = threadIdx.x & 31;
  const int waveId = threadIdx.x >> 5;
  const int tile   = blockIdx.x * (blockDim.x >> 5) + waveId;
  const int b0     = tile * 16;
  if (b0 >= B) return;  // wave-uniform

  const int  dir   = blockIdx.y;           // 0 = forward, 1 = backward
  const bool rev   = (dir != 0);
  const float* Wih = rev ? WihB : WihF;
  const float* Whh = rev ? WhhB : WhhF;
  const float* bih = rev ? bihB : bihF;
  const float* bhh = rev ? bhhB : bhhF;
  const int outOff = dir * H;

  const int n    = lane & 15;   // batch column (B/D frags) == M row (A frags)
  const int hi   = lane >> 4;   // lane half
  const int bRow = b0 + n;

  // ---- Load weights as WMMA A-fragments (constant across time) ----
  // A 16x32 f16 layout: lane holds M = lane%16;
  //   element e -> K = 16*hi + e (contiguous order chosen here, same as B; not the ISA table order)
  v16h  Aih[NT][KTX];
  v16h  Ahh[NT];
  float bias[NT][8];

#pragma unroll
  for (int mt = 0; mt < NT; ++mt) {
    const int  m     = mt * 16 + n;       // padded gate row
    const int  gate  = m / Hp;
    const int  hidp  = m % Hp;
    const bool mval  = (hidp < H);
    const int  morig = gate * H + hidp;   // row in the original [4H x *] weight
#pragma unroll
    for (int kt = 0; kt < KTX; ++kt) {
#pragma unroll
      for (int e = 0; e < 16; ++e) {
        const int k = kt * 32 + hi * 16 + e;
        Aih[mt][kt][e] = (_Float16)(mval ? Wih[morig * I + k] : 0.f);
      }
    }
#pragma unroll
    for (int e = 0; e < 16; ++e) {
      const int k = hi * 16 + e;  // padded hidden idx
      Ahh[mt][e] = (_Float16)((mval && k < H) ? Whh[morig * H + k] : 0.f);
    }
#pragma unroll
    for (int r = 0; r < 8; ++r) {
      const int mm = mt * 16 + r + 8 * hi;  // D-layout gate row
      const int g2 = mm / Hp, h2 = mm % Hp;
      bias[mt][r] = (h2 < H) ? (bih[g2 * H + h2] + bhh[g2 * H + h2]) : 0.f;
    }
  }

  // ---- Recurrent state (all registers) ----
  v8f cst[CT];
  v8f ht[2];  // h^T tiles (D layout); ht[1] stays 0 when CT==1
#pragma unroll
  for (int ct = 0; ct < CT; ++ct)
#pragma unroll
    for (int r = 0; r < 8; ++r) cst[ct][r] = 0.f;
#pragma unroll
  for (int r = 0; r < 8; ++r) { ht[0][r] = 0.f; ht[1][r] = 0.f; }
  v16h Bh;
#pragma unroll
  for (int e = 0; e < 16; ++e) Bh[e] = (_Float16)0.f;

  for (int s = 0; s < T; ++s) {
    const int t = rev ? (T - 1 - s) : s;

    // ---- x^T B-fragment: lane half hi reads 16 contiguous features ----
    // B 32x16 f16 layout: lane holds N = lane%16; element e -> K = 16*hi + e (same contiguous order as A, not the ISA table order)
    v16h Bx[KTX];
    const float* xrow = x + ((size_t)bRow * T + t) * I;
#pragma unroll
    for (int kt = 0; kt < KTX; ++kt) {
      const float4* xq = reinterpret_cast<const float4*>(xrow + kt * 32 + hi * 16);
      const float4 q0 = xq[0], q1 = xq[1], q2 = xq[2], q3 = xq[3];
      const float xv[16] = {q0.x, q0.y, q0.z, q0.w, q1.x, q1.y, q1.z, q1.w,
                            q2.x, q2.y, q2.z, q2.w, q3.x, q3.y, q3.z, q3.w};
#pragma unroll
      for (int e = 0; e < 16; ++e) Bx[kt][e] = (_Float16)xv[e];
    }

    // ---- gates^T = bias + Wih*x^T + Whh*h^T ----
    v8f gacc[NT];
#pragma unroll
    for (int mt = 0; mt < NT; ++mt) {
      v8f acc;
#pragma unroll
      for (int r = 0; r < 8; ++r) acc[r] = bias[mt][r];
#pragma unroll
      for (int kt = 0; kt < KTX; ++kt) acc = wmma_f16f32(Aih[mt][kt], Bx[kt], acc);
      acc = wmma_f16f32(Ahh[mt], Bh, acc);
      gacc[mt] = acc;
    }

    // ---- elementwise LSTM cell (per lane = per batch column) ----
#pragma unroll
    for (int ct = 0; ct < CT; ++ct) {
#pragma unroll
      for (int r = 0; r < 8; ++r) {
        const float iv = sigm(gacc[0 * CT + ct][r]);
        const float fv = sigm(gacc[1 * CT + ct][r]);
        const float gv = tanh_fast(gacc[2 * CT + ct][r]);
        const float ov = sigm(gacc[3 * CT + ct][r]);
        const float cv = fv * cst[ct][r] + iv * gv;
        cst[ct][r]     = cv;
        ht[ct][r]      = ov * tanh_fast(cv);
      }
    }

    // ---- store h (f32) into the next layer's input buffer ----
    float* orow = out + ((size_t)bRow * T + t) * outCols + outOff;
#pragma unroll
    for (int ct = 0; ct < CT; ++ct) {
      const int base = ct * 16 + 8 * hi;
      if (base < H) {
        const v8f h = ht[ct];
        *reinterpret_cast<float4*>(orow + base)     = make_float4(h[0], h[1], h[2], h[3]);
        *reinterpret_cast<float4*>(orow + base + 4) = make_float4(h[4], h[5], h[6], h[7]);
      }
    }

    // ---- rebuild h^T B-fragment for the next step ----
    // Need h[K = 16*hi + e]; the opposite lane-half's rows via shfl_xor(16).
    float hsw0[8], hsw1[8];
#pragma unroll
    for (int r = 0; r < 8; ++r) hsw0[r] = __shfl_xor(ht[0][r], 16, 32);
#pragma unroll
    for (int r = 0; r < 8; ++r)
      hsw1[r] = (CT > 1) ? __shfl_xor(ht[1][r], 16, 32) : 0.f;
#pragma unroll
    for (int e = 0; e < 16; ++e) {
      float hv;
      if (e < 8) hv = hi ? hsw1[e]      : ht[0][e];
      else       hv = hi ? ht[1][e - 8] : hsw0[e - 8];
      Bh[e] = (_Float16)hv;
    }
  }
}

// y[b, o] = out4[b, 0, :] . W[o, :] + bias[o]
__global__ __launch_bounds__(256)
void final_proj_kernel(const float* __restrict__ x4,  // [B, T, 16]
                       const float* __restrict__ W,   // [24, 16]
                       const float* __restrict__ bfin,
                       float* __restrict__ y, int B, int T) {
  const int idx = blockIdx.x * blockDim.x + threadIdx.x;
  if (idx >= B * 24) return;
  const int b = idx / 24, o = idx % 24;
  const float* xr = x4 + (size_t)b * T * 16;  // t = 0
  float acc = bfin[o];
#pragma unroll
  for (int k = 0; k < 16; ++k) acc += xr[k] * W[o * 16 + k];
  y[idx] = acc;
}

extern "C" void kernel_launch(void* const* d_in, const int* in_sizes, int n_in,
                              void* d_out, int out_size, void* d_ws, size_t ws_size,
                              hipStream_t stream) {
  (void)out_size; (void)ws_size;
  const int B = 2048, T = 256;

  const float* x0 = (const float*)d_in[0];  // [2048, 256, 32]
  const float* W4[4][8];  // [layer][WihF, WhhF, bihF, bhhF, WihB, WhhB, bihB, bhhB]
  const float* Wfin = nullptr;
  const float* bfin = nullptr;

  if (n_in >= 35 && in_sizes[1] == 384) {
    // jax-pytree sorted-key flatten: inputs, W, b, l1{b(4), f(4)}, ..., l4
    Wfin = (const float*)d_in[1];
    bfin = (const float*)d_in[2];
    for (int L = 0; L < 4; ++L) {
      const int base = 3 + L * 8;
      for (int j = 0; j < 4; ++j) W4[L][4 + j] = (const float*)d_in[base + j];     // 'b'
      for (int j = 0; j < 4; ++j) W4[L][j]     = (const float*)d_in[base + 4 + j]; // 'f'
    }
  } else {
    // dict insertion order: inputs, l1{f(4), b(4)}, ..., l4, W, b
    for (int L = 0; L < 4; ++L) {
      const int base = 1 + L * 8;
      for (int j = 0; j < 8; ++j) W4[L][j] = (const float*)d_in[base + j];
    }
    Wfin = (const float*)d_in[33];
    bfin = (const float*)d_in[34];
  }

  // Workspace ping-pong: bufA = 2048*256*64 f32 (134 MB), bufB = 2048*256*32 f32
  float* bufA = (float*)d_ws;
  float* bufB = (float*)((char*)d_ws + (size_t)B * T * 64 * sizeof(float));

  dim3 blk(256, 1, 1);
  dim3 grd(B / (16 * 8), 2, 1);  // 16 blocks x 8 waves x 16 rows = 2048; y = dir

  lstm_layer_kernel<32, 32, 32><<<grd, blk, 0, stream>>>(
      x0, bufA, 64, W4[0][0], W4[0][1], W4[0][2], W4[0][3],
      W4[0][4], W4[0][5], W4[0][6], W4[0][7], B, T);
  lstm_layer_kernel<64, 16, 16><<<grd, blk, 0, stream>>>(
      bufA, bufB, 32, W4[1][0], W4[1][1], W4[1][2], W4[1][3],
      W4[1][4], W4[1][5], W4[1][6], W4[1][7], B, T);
  lstm_layer_kernel<32, 16, 16><<<grd, blk, 0, stream>>>(
      bufB, bufA, 32, W4[2][0], W4[2][1], W4[2][2], W4[2][3],
      W4[2][4], W4[2][5], W4[2][6], W4[2][7], B, T);
  lstm_layer_kernel<32, 8, 16><<<grd, blk, 0, stream>>>(
      bufA, bufB, 16, W4[3][0], W4[3][1], W4[3][2], W4[3][3],
      W4[3][4], W4[3][5], W4[3][6], W4[3][7], B, T);

  final_proj_kernel<<<(B * 24 + 255) / 256, 256, 0, stream>>>(
      bufB, Wfin, bfin, (float*)d_out, B, T);
}
